// Mlp_91044716740733
// MI455X (gfx1250) — 8-bit matrix operand order repaired and checked by reading, compiled, not run on hardware
//
#include <hip/hip_runtime.h>


typedef int v8i __attribute__((ext_vector_type(8)));
typedef _Float16 h8v __attribute__((ext_vector_type(8)));
typedef int gv4i __attribute__((vector_size(16)));

#define QP  127.0f
#define EPS 1e-6f

#define AS1 __attribute__((address_space(1)))
#define AS3 __attribute__((address_space(3)))

#ifndef __has_builtin
#define __has_builtin(x) 0
#endif

#if __has_builtin(__builtin_amdgcn_global_load_async_to_lds_b128)
#define HAVE_ASYNC_LDS 1
#else
#define HAVE_ASYNC_LDS 0
#endif

__device__ __forceinline__ void async_copy_b128(const signed char* g, signed char* l) {
#if HAVE_ASYNC_LDS
  __builtin_amdgcn_global_load_async_to_lds_b128((AS1 gv4i*)g, (AS3 gv4i*)l,
                                                  0,  0);
#else
  *(int4*)l = *(const int4*)g;
#endif
}

__device__ __forceinline__ void wait_async_zero() {
#if HAVE_ASYNC_LDS
#if __has_builtin(__builtin_amdgcn_s_wait_asynccnt)
  __builtin_amdgcn_s_wait_asynccnt(0);
#else
  asm volatile("s_wait_asynccnt 0x0" ::: "memory");
#endif
#endif
}

__global__ void init_scales(unsigned* s) {
  if (threadIdx.x < 4) s[threadIdx.x] = 0u;
}

__global__ void amax_f32(const float* __restrict__ x, size_t n4, unsigned* slot) {
  size_t i = (size_t)blockIdx.x * blockDim.x + threadIdx.x;
  size_t stride = (size_t)gridDim.x * blockDim.x;
  float m = 0.f;
  const float4* x4 = (const float4*)x;
  for (; i < n4; i += stride) {
    float4 v = x4[i];
    m = fmaxf(m, fmaxf(fmaxf(fabsf(v.x), fabsf(v.y)),
                       fmaxf(fabsf(v.z), fabsf(v.w))));
  }
  for (int off = 16; off > 0; off >>= 1) m = fmaxf(m, __shfl_xor(m, off, 32));
  __shared__ float red[8];
  int lane = threadIdx.x & 31, wave = threadIdx.x >> 5;
  if (lane == 0) red[wave] = m;
  __syncthreads();
  if (threadIdx.x == 0) {
    float mm = red[0];
    for (int w = 1; w < (int)(blockDim.x >> 5); ++w) mm = fmaxf(mm, red[w]);
    atomicMax(slot, __float_as_uint(mm));
  }
}

__global__ void quant_f32_i8(const float* __restrict__ x, signed char* __restrict__ q,
                             const unsigned* __restrict__ scales, int slot, size_t n4) {
  float s = fmaxf(__uint_as_float(scales[slot]), EPS) / QP;
  float inv = 1.f / s;
  size_t i = (size_t)blockIdx.x * blockDim.x + threadIdx.x;
  size_t stride = (size_t)gridDim.x * blockDim.x;
  const float4* x4 = (const float4*)x;
  unsigned* q4 = (unsigned*)q;
  for (; i < n4; i += stride) {
    float4 v = x4[i];
    int a = (int)rintf(fminf(fmaxf(v.x * inv, -QP), QP));
    int b = (int)rintf(fminf(fmaxf(v.y * inv, -QP), QP));
    int c = (int)rintf(fminf(fmaxf(v.z * inv, -QP), QP));
    int d = (int)rintf(fminf(fmaxf(v.w * inv, -QP), QP));
    q4[i] = (a & 0xff) | ((b & 0xff) << 8) | ((c & 0xff) << 16) | ((d & 0xff) << 24);
  }
}

__global__ void quant_f16_i8(const _Float16* __restrict__ h, signed char* __restrict__ q,
                             const unsigned* __restrict__ scales, int slot, size_t n8) {
  float s = fmaxf(__uint_as_float(scales[slot]), EPS) / QP;
  float inv = 1.f / s;
  size_t i = (size_t)blockIdx.x * blockDim.x + threadIdx.x;
  size_t stride = (size_t)gridDim.x * blockDim.x;
  const h8v* hv = (const h8v*)h;
  uint2* q8 = (uint2*)q;
  for (; i < n8; i += stride) {
    h8v v = hv[i];
    unsigned lo = 0, hi = 0;
#pragma unroll
    for (int j = 0; j < 4; ++j) {
      int b = (int)rintf(fminf(fmaxf((float)v[j] * inv, -QP), QP));
      lo |= (unsigned)(b & 0xff) << (8 * j);
    }
#pragma unroll
    for (int j = 0; j < 4; ++j) {
      int b = (int)rintf(fminf(fmaxf((float)v[4 + j] * inv, -QP), QP));
      hi |= (unsigned)(b & 0xff) << (8 * j);
    }
    q8[i] = make_uint2(lo, hi);
  }
}

template <bool GELU>
__global__ __launch_bounds__(256) void gemm_i8_wmma(
    const signed char* __restrict__ A,
    const signed char* __restrict__ Bm,
    const float* __restrict__ bias,
    const unsigned* __restrict__ scales, int slotA, int slotB,
    _Float16* __restrict__ Hout,
    float* __restrict__ Fout,
    unsigned* __restrict__ amaxSlot,
    int N, int K) {
  constexpr int LP = 80;
  __shared__ __align__(16) signed char sA[2][256 * LP];
  __shared__ __align__(16) signed char sB[2][64 * LP];
  __shared__ float red[8];

  const int tid = threadIdx.x;
  const int lane = tid & 31;
  const int wave = tid >> 5;
  const int hs = lane >> 4;
  const int ln = lane & 15;
  const int wm = wave >> 1;
  const int wn = wave & 1;
  const int blockM = blockIdx.y * 256;
  const int blockN = blockIdx.x * 64;

  const float sa = fmaxf(__uint_as_float(scales[slotA]), EPS) / QP;
  const float sb = fmaxf(__uint_as_float(scales[slotB]), EPS) / QP;
  const float sAB = sa * sb;

  const signed char* aRow = A + (size_t)(blockM + tid) * K;
  const signed char* bRow = Bm + (size_t)(blockN + (tid >> 2)) * K + (tid & 3) * 16;

  auto stage = [&](int kk, int buf) {
    signed char* la = &sA[buf][tid * LP];
    signed char* lb = &sB[buf][(tid >> 2) * LP + (tid & 3) * 16];
    const signed char* ga = aRow + kk;
#pragma unroll
    for (int j = 0; j < 4; ++j) async_copy_b128(ga + j * 16, la + j * 16);
    async_copy_b128(bRow + kk, lb);
    if (kk + 64 < K) {
      __builtin_prefetch(aRow + kk + 64, 0, 3);
      __builtin_prefetch(bRow + kk + 64, 0, 3);
    }
  };

  v8i acc[4][2] = {};

  const int nslab = K >> 6;
  stage(0, 0);
  for (int i = 0; i < nslab; ++i) {
    const int b = i & 1;
    wait_async_zero();
    __syncthreads();
    if (i + 1 < nslab) stage((i + 1) << 6, b ^ 1);

    v8i af[4];
#pragma unroll
    for (int mt = 0; mt < 4; ++mt) {
      const signed char* base = &sA[b][(wm * 64 + mt * 16 + ln) * LP + hs * 16];
      int2 p0 = *(const int2*)(base);
      int2 p1 = *(const int2*)(base + 8);
      int2 p2 = *(const int2*)(base + 32);
      int2 p3 = *(const int2*)(base + 40);
      v8i t = {p0.x, p0.y, p1.x, p1.y, p2.x, p2.y, p3.x, p3.y};
      af[mt] = t;
    }
#pragma unroll
    for (int nt = 0; nt < 2; ++nt) {
      const signed char* bbase = &sB[b][(wn * 32 + nt * 16 + ln) * LP + hs * 16];
      int4 q0 = *(const int4*)(bbase);
      int4 q1 = *(const int4*)(bbase + 32);
      v8i bf = {q0.x, q0.y, q0.z, q0.w, q1.x, q1.y, q1.z, q1.w};
#pragma unroll
      for (int mt = 0; mt < 4; ++mt)
        acc[mt][nt] = __builtin_amdgcn_wmma_i32_16x16x64_iu8(
            true, af[mt], true, bf, acc[mt][nt], false, false);
    }
  }

  float bi[2];
#pragma unroll
  for (int nt = 0; nt < 2; ++nt) bi[nt] = bias[blockN + wn * 32 + nt * 16 + ln];

  float smax = 0.f;
#pragma unroll
  for (int mt = 0; mt < 4; ++mt) {
#pragma unroll
    for (int nt = 0; nt < 2; ++nt) {
      const int n = blockN + wn * 32 + nt * 16 + ln;
#pragma unroll
      for (int g = 0; g < 8; ++g) {
        const int m = blockM + wm * 64 + mt * 16 + hs * 8 + g;
        float f = (float)acc[mt][nt][g] * sAB + bi[nt];
        if constexpr (GELU) {
          float ge = 0.5f * f * (1.0f + erff(f * 0.7071067811865475f));
          Hout[(size_t)m * N + n] = (_Float16)ge;
          smax = fmaxf(smax, fabsf(ge));
        } else {
          Fout[(size_t)m * N + n] = f;
        }
      }
    }
  }

  if constexpr (GELU) {
    for (int off = 16; off > 0; off >>= 1) smax = fmaxf(smax, __shfl_xor(smax, off, 32));
    if (lane == 0) red[wave] = smax;
    __syncthreads();
    if (tid == 0) {
      float mm = red[0];
      for (int w = 1; w < 8; ++w) mm = fmaxf(mm, red[w]);
      atomicMax(amaxSlot, __float_as_uint(mm));
    }
  }
}

extern "C" void kernel_launch(void* const* d_in, const int* in_sizes, int n_in,
                              void* d_out, int out_size, void* d_ws, size_t ws_size,
                              hipStream_t stream) {
  (void)in_sizes; (void)n_in; (void)out_size; (void)ws_size;
  const float* x  = (const float*)d_in[0];
  const float* w1 = (const float*)d_in[1];
  const float* b1 = (const float*)d_in[2];
  const float* w2 = (const float*)d_in[3];
  const float* b2 = (const float*)d_in[4];
  float* y = (float*)d_out;

  const size_t M   = 32768;
  const size_t nX  = M * 1024;
  const size_t nW1 = 4096ull * 1024;
  const size_t nW2 = 1024ull * 4096;
  const size_t nH  = M * 4096;

  unsigned* scales   = (unsigned*)d_ws;
  signed char* xq    = (signed char*)d_ws + 256;
  signed char* w1q   = xq + nX;
  signed char* w2q   = w1q + nW1;
  signed char* hq    = w2q + nW2;
  _Float16* h16      = (_Float16*)(hq + nH);

  init_scales<<<1, 32, 0, stream>>>(scales);

  amax_f32<<<1024, 256, 0, stream>>>(x,  nX  / 4, scales + 0);
  amax_f32<<<256,  256, 0, stream>>>(w1, nW1 / 4, scales + 1);
  amax_f32<<<256,  256, 0, stream>>>(w2, nW2 / 4, scales + 2);

  quant_f32_i8<<<1024, 256, 0, stream>>>(x,  xq,  scales, 0, nX  / 4);
  quant_f32_i8<<<256,  256, 0, stream>>>(w1, w1q, scales, 1, nW1 / 4);
  quant_f32_i8<<<256,  256, 0, stream>>>(w2, w2q, scales, 2, nW2 / 4);

  dim3 g1(4096 / 64, 32768 / 256);
  gemm_i8_wmma<true><<<g1, 256, 0, stream>>>(xq, w1q, b1, scales, 0, 1,
                                             h16, nullptr, scales + 3, 4096, 1024);

  quant_f16_i8<<<2048, 256, 0, stream>>>(h16, hq, scales, 3, nH / 8);

  dim3 g2(1024 / 64, 32768 / 256);
  gemm_i8_wmma<false><<<g2, 256, 0, stream>>>(hq, w2q, b2, scales, 3, 2,
                                              nullptr, y, nullptr, 1024, 4096);
}
